// NTXentLoss_84104049590874
// MI455X (gfx1250) — matrix operand order repaired and checked by reading, not compiled, not run on hardware
//
#include <hip/hip_runtime.h>
#include <hip/hip_fp16.h>

typedef _Float16 v16h __attribute__((ext_vector_type(16)));
typedef _Float16 v8h  __attribute__((ext_vector_type(8)));
typedef float    v8f  __attribute__((ext_vector_type(8)));
typedef unsigned int u32x4 __attribute__((ext_vector_type(4)));
typedef int          i32x4 __attribute__((ext_vector_type(4)));
typedef int          i32x8 __attribute__((ext_vector_type(8)));

#define N_ROWS 4096
#define D      512
#define TWO_N  8192
#define NT     (TWO_N / 16)   // 512 column tiles
#define INV_T  2.0f           // 1 / TEMPERATURE

#if __has_builtin(__builtin_amdgcn_tensor_load_to_lds)
#define USE_TDM 1
#else
#define USE_TDM 0
#endif

// ---------------------------------------------------------------------------
// Kernel 1: row-normalize concat(z_i, z_j) -> f16 zn[8192][512] in workspace
// ---------------------------------------------------------------------------
__global__ __launch_bounds__(256)
void normalize_rows(const float* __restrict__ zi, const float* __restrict__ zj,
                    _Float16* __restrict__ zn) {
  int row = blockIdx.x;
  const float* src = (row < N_ROWS) ? (zi + (size_t)row * D)
                                    : (zj + (size_t)(row - N_ROWS) * D);
  int t = threadIdx.x;                       // 256 threads, 2 elems each
  float2 v = ((const float2*)src)[t];
  __shared__ float red[256];
  red[t] = v.x * v.x + v.y * v.y;
  __syncthreads();
  for (int s = 128; s > 0; s >>= 1) {
    if (t < s) red[t] += red[t + s];
    __syncthreads();
  }
  float inv = 1.0f / fmaxf(sqrtf(red[0]), 1e-8f);
  _Float16* dst = zn + (size_t)row * D;
  dst[2 * t]     = (_Float16)(v.x * inv);
  dst[2 * t + 1] = (_Float16)(v.y * inv);
}

// ---------------------------------------------------------------------------
// Kernel 2: fused sim GEMM (WMMA f16) + masked exp-sum + positive capture.
// 128 blocks x 128 threads (4 waves). Wave w owns rows [rowBase, rowBase+16).
// B panels (16 cols x 512 K = 16KB) staged into double-buffered LDS by the
// Tensor Data Mover (one tensor_load_to_lds per panel, TENSORcnt-tracked),
// shared by the 4 waves of the block.
// ---------------------------------------------------------------------------
__global__ __launch_bounds__(128)
void ntxent_wmma(const _Float16* __restrict__ zn, float* __restrict__ partials) {
  __shared__ _Float16 Bs[2][16 * D];         // 2 x 16 KB

  const int tid     = threadIdx.x;           // 0..127
  const int lane    = tid & 31;
  const int wave    = tid >> 5;
  const int lane16  = lane & 15;
  const int laneHi  = lane >> 4;             // 0: lanes 0-15, 1: lanes 16-31
  const int rowBase = (blockIdx.x * 4 + wave) * 16;

  // ---- A fragments: 16 rows x 512 K, kept in registers for all 512 tiles.
  // A k order used here (contiguous, not ISA table): lane group 0 holds K k0..k0+15,
  // lane group 1 holds K k0+16..k0+31 (matches B); row = lane16.
  const _Float16* aRow = zn + (size_t)(rowBase + lane16) * D;
  v16h a[16];
#pragma unroll
  for (int i = 0; i < 16; ++i) {
    int k0 = i * 32 + laneHi * 16;
    union { v16h v; v8h h[2]; } u;
    u.h[0] = *(const v8h*)(aRow + k0);
    u.h[1] = *(const v8h*)(aRow + k0 + 8);
    a[i] = u.v;
  }

  float s[8], p[8];                          // per-lane sumexp / positive logit
#pragma unroll
  for (int r = 0; r < 8; ++r) { s[r] = 0.0f; p[r] = 0.0f; }

#if USE_TDM
  // ---- Tensor DMA descriptor pieces (uniform -> SGPRs). D# per ISA §8:
  // group0: count=1, lds_addr[63:32], global_addr[120:64], type=2 [127:126]
  // group1: data_size=2B, tensor_dim0=512, tensor_dim1=8192, tile=512x16,
  //         tensor_dim0_stride=512.
  const unsigned ldsBase = (unsigned)(size_t)(&Bs[0][0]);
  #define TDM_ISSUE(panel, buf)                                                \
    {                                                                          \
      unsigned long long ga =                                                  \
          (unsigned long long)(size_t)(zn + (size_t)(panel) * 16 * D);         \
      u32x4 g0;                                                                \
      g0[0] = 1u;                                  /* count=1 */               \
      g0[1] = ldsBase + (unsigned)(buf) * (16 * D * 2); /* lds_addr */         \
      g0[2] = (unsigned)(ga & 0xFFFFFFFFu);                                    \
      g0[3] = (unsigned)((ga >> 32) & 0x01FFFFFFu) | (2u << 30); /* type=2 */  \
      i32x8 g1;                                                                \
      g1[0] = 0x00010000;        /* wg_mask=0, data_size=1 (2 bytes) */        \
      g1[1] = (int)(D << 16);    /* tensor_dim0 = 512 (bits 79:48) */          \
      g1[2] = (int)(TWO_N << 16);/* tensor_dim1 = 8192 (bits 111:80) */        \
      g1[3] = (int)(D << 16);    /* tile_dim0 = 512 (bits 127:112) */          \
      g1[4] = 16;                /* tile_dim1 = 16, tile_dim2 = 0 */           \
      g1[5] = (int)D;            /* tensor_dim0_stride = 512 */                \
      g1[6] = 0; g1[7] = 0;                                                    \
      i32x4 gz = {0, 0, 0, 0};                                                 \
      _Pragma("clang diagnostic push")                                         \
      TDM_CALL(g0, g1, gz, gz);                                                \
      _Pragma("clang diagnostic pop")                                          \
    }
  #if defined(__clang_major__) && __clang_major__ >= 23
    #define TDM_CALL(g0, g1, g2, g3)                                           \
      { i32x8 g4 = {0,0,0,0,0,0,0,0};                                          \
        __builtin_amdgcn_tensor_load_to_lds(g0, g1, g2, g3, g4, 0); }
  #else
    #define TDM_CALL(g0, g1, g2, g3)                                           \
      __builtin_amdgcn_tensor_load_to_lds(g0, g1, g2, g3, 0);
  #endif

  if (wave == 0) { TDM_ISSUE(0, 0); }        // prologue: panel 0 -> buf 0
#else
  {                                          // fallback: cooperative copy
    const int4* g = (const int4*)zn;
    int4* l = (int4*)(&Bs[0][0]);
#pragma unroll
    for (int j = 0; j < 8; ++j) l[tid + 128 * j] = g[tid + 128 * j];
  }
  __syncthreads();
#endif

  for (int ct = 0; ct < NT; ++ct) {
    const int cur = ct & 1;

#if USE_TDM
    // Prefetch panel ct+1 into the other buffer (freed by last iteration's
    // trailing barrier), then ensure panel ct has landed.
    if (wave == 0) {
      if (ct + 1 < NT) {
        TDM_ISSUE(ct + 1, cur ^ 1);
        __builtin_amdgcn_s_wait_tensorcnt(1);   // panel ct complete
      } else {
        __builtin_amdgcn_s_wait_tensorcnt(0);
      }
    }
    __syncthreads();                            // panel ct visible to all waves
#else
    if (ct + 1 < NT) {
      const int4* g = (const int4*)(zn + (size_t)(ct + 1) * 16 * D);
      int4* l = (int4*)(&Bs[cur ^ 1][0]);
#pragma unroll
      for (int j = 0; j < 8; ++j) l[tid + 128 * j] = g[tid + 128 * j];
    }
#endif

    // B k order used here (contiguous, not ISA table): lane = column, lanes 0-15 hold K k0..k0+15,
    // lanes 16-31 hold K k0+16..k0+31 -> contiguous 32B per lane from panel.
    const _Float16* bBase = &Bs[cur][lane16 * D + laneHi * 16];

    v8f acc0 = {}, acc1 = {};                  // two chains -> 2 WMMAs in flight
#pragma unroll
    for (int i = 0; i < 16; i += 2) {
      v16h b0 = *(const v16h*)(bBase + i * 32);
      v16h b1 = *(const v16h*)(bBase + (i + 1) * 32);
      acc0 = __builtin_amdgcn_wmma_f32_16x16x32_f16(
          false, a[i], false, b0, (short)0, acc0, false, false);
      acc1 = __builtin_amdgcn_wmma_f32_16x16x32_f16(
          false, a[i + 1], false, b1, (short)0, acc1, false, false);
    }

    // C layout: lane = column (within half), VGPR r -> row r (lo) / r+8 (hi)
    const int col = ct * 16 + lane16;
#pragma unroll
    for (int r = 0; r < 8; ++r) {
      int row = rowBase + r + laneHi * 8;
      float v = INV_T * (acc0[r] + acc1[r]);
      if (col != row)            s[r] += __expf(v);   // diagonal masked out
      if (col == (row ^ N_ROWS)) p[r]  = v;           // positive pair logit
    }
    __syncthreads();                           // release buf for TDM overwrite
  }

  // ---- reduce across the 16 lanes of each half (columns of each row)
  float wsum = 0.0f;
#pragma unroll
  for (int r = 0; r < 8; ++r) {
#pragma unroll
    for (int off = 1; off <= 8; off <<= 1) {
      s[r] += __shfl_xor(s[r], off, 32);
      p[r] += __shfl_xor(p[r], off, 32);
    }
    wsum += __logf(s[r]) - p[r];               // logsumexp - positive
  }
  wsum += __shfl_xor(wsum, 16, 32);            // combine lane halves
  if (lane == 0) partials[blockIdx.x * 4 + wave] = wsum;
}

// ---------------------------------------------------------------------------
// Kernel 3: deterministic final reduction of 512 wave partials -> loss scalar
// loss = sum_{8192 rows}(logsumexp - pos) / (2N)
// ---------------------------------------------------------------------------
__global__ __launch_bounds__(256)
void final_reduce(const float* __restrict__ partials, float* __restrict__ out) {
  __shared__ float red[256];
  int t = threadIdx.x;
  red[t] = partials[t] + partials[t + 256];
  __syncthreads();
  for (int s = 128; s > 0; s >>= 1) {
    if (t < s) red[t] += red[t + s];
    __syncthreads();
  }
  if (t == 0) out[0] = red[0] * (1.0f / (float)TWO_N);
}

extern "C" void kernel_launch(void* const* d_in, const int* in_sizes, int n_in,
                              void* d_out, int out_size, void* d_ws, size_t ws_size,
                              hipStream_t stream) {
  const float* zi = (const float*)d_in[0];
  const float* zj = (const float*)d_in[1];
  _Float16* zn       = (_Float16*)d_ws;                               // 8 MB
  float*    partials = (float*)((char*)d_ws + (size_t)TWO_N * D * sizeof(_Float16));

  normalize_rows<<<TWO_N, 256, 0, stream>>>(zi, zj, zn);
  ntxent_wmma<<<TWO_N / 64, 128, 0, stream>>>(zn, partials);
  final_reduce<<<1, 256, 0, stream>>>(partials, (float*)d_out);
}
